// RecurrentHierarchicalTransformer_83519934038337
// MI455X (gfx1250) — compile-verified, matrix operand order checked by reading, not run on hardware
//
#include <hip/hip_runtime.h>
#include <hip/hip_bf16.h>
#include <math.h>

// ---------------------------------------------------------------------------
// Types
// ---------------------------------------------------------------------------
typedef __bf16 bf16_t;
typedef __attribute__((ext_vector_type(8)))  bf16_t v8bf;
typedef __attribute__((ext_vector_type(16))) bf16_t v16bf;
typedef __attribute__((ext_vector_type(8)))  float  v8f;

static __device__ __forceinline__ bf16_t f2bf(float f) {
  union { float f; unsigned u; } in; in.f = f;
  unsigned u = in.u;
  unsigned r = u + 0x7FFFu + ((u >> 16) & 1u);   // round-to-nearest-even
  union { unsigned short s; bf16_t b; } out;
  out.s = (unsigned short)(r >> 16);
  return out.b;
}

// ---------------------------------------------------------------------------
// bf16 WMMA GEMM:  C[M,N] = A[M,K] * B[K,N]  (+bias +gelu +residual)
//   - block = 256 threads = 8 waves; block tile 128x64; wave tile 32x32
//     (4 x v_wmma_f32_16x16x32_bf16 per K-step per wave, A/B fragments each
//      reused twice -> 2.0 ds_load_b128 per WMMA)
//   - software pipelined: tile t+1 is loaded from global into registers while
//     tile t's WMMAs execute between the two workgroup barriers; tile t+2 is
//     prefetched (global_prefetch_b8)
//   - A: bf16 row-major (lda), per-batch offset = b*Ab + h*Ah
//   - B: bf16; BT==false -> row-major KxN (ldb); BT==true -> pointer is B^T
//        stored NxK row-major (ldb = its row stride)
//   - C element (r,c) at cbase + r*ldc + c*cstride, fp32 and/or bf16 stores
//   M % 128 == 0, N % 64 == 0, K % 32 == 0 for every GEMM in this model, so
//   no bounds guards and EXEC is all-ones at every v_wmma (ISA requirement).
// ---------------------------------------------------------------------------
template<bool BT>
__global__ __launch_bounds__(256)
void gemm_wmma_bf16(const bf16_t* __restrict__ A, int lda, long long Ab, long long Ah,
                    const bf16_t* __restrict__ Bm, int ldb, long long Bb, long long Bh,
                    float* __restrict__ Cf, bf16_t* __restrict__ Cb,
                    int ldc, int cstride, long long Cbs, long long Chs,
                    const float* __restrict__ bias, const float* __restrict__ resid,
                    int gelu, int K)
{
  __shared__ bf16_t As[128][32];   // 8 KB
  __shared__ bf16_t Bs[64][32];    // 4 KB, N-major: Bs[n][k]

  const int tid  = threadIdx.x;
  const int bb   = blockIdx.z >> 3;     // batch (b)
  const int hh   = blockIdx.z & 7;      // head  (h)
  const long long abase = (long long)bb * Ab + (long long)hh * Ah;
  const long long bbase = (long long)bb * Bb + (long long)hh * Bh;
  const long long cbase = (long long)bb * Cbs + (long long)hh * Chs;

  const int m0 = blockIdx.y * 128;
  const int n0 = blockIdx.x * 64;

  const int w    = tid >> 5;
  const int lane = tid & 31;
  const int wr   = w & 3;        // wave row   (4 x 32 rows)
  const int wc   = w >> 2;       // wave col   (2 x 32 cols)
  const int hf   = lane >> 4;    // lane half
  const int lrow = lane & 15;
  // 16-bit A-matrix VGPR layout (ISA 7.12.2): lanes 0-15 hold K{0..7,16..23},
  // lanes 16-31 hold K{8..15,24..31}
  const int kb0 = hf * 8;
  const int kb1 = 16 + hf * 8;

  v8f c00 = {}, c01 = {}, c10 = {}, c11 = {};

  // A tile loader: 128x32, each thread stages 16 elems (two b128)
  const int ar = tid >> 1;             // 0..127
  const int ac = (tid & 1) * 16;       // 0 or 16
  // B tile loader (BT): 64x32, one b128 per thread
  const int br  = tid >> 2;            // 0..63
  const int bc8 = (tid & 3) * 8;       // 0,8,16,24
  // B tile loader (!BT): contiguous along N, transposed into LDS
  const int bk  = tid >> 3;            // 0..31
  const int bn8 = (tid & 7) * 8;       // 0..56

  const bf16_t* Ag = A + abase + (long long)(m0 + ar) * lda + ac;
  const bf16_t* Bg = BT ? (Bm + bbase + (long long)(n0 + br) * ldb + bc8)
                        : (Bm + bbase + (long long)bk * ldb + n0 + bn8);
  const long long bstep = BT ? 32 : (long long)32 * ldb;

  // stage tile 0
  v8bf pa0 = *(const v8bf*)(Ag);
  v8bf pa1 = *(const v8bf*)(Ag + 8);
  v8bf pb  = *(const v8bf*)(Bg);

  for (int kt = 0; kt < K; kt += 32) {
    // --- commit staged tile to LDS
    *(v8bf*)(&As[ar][ac])     = pa0;
    *(v8bf*)(&As[ar][ac + 8]) = pa1;
    if (BT) {
      *(v8bf*)(&Bs[br][bc8]) = pb;
    } else {
      #pragma unroll
      for (int e = 0; e < 8; ++e) Bs[bn8 + e][bk] = pb[e];
    }
    __syncthreads();

    // --- stage tile t+1 from global (overlaps with WMMA below), prefetch t+2
    if (kt + 32 < K) {
      Ag += 32;
      Bg += bstep;
      pa0 = *(const v8bf*)(Ag);
      pa1 = *(const v8bf*)(Ag + 8);
      pb  = *(const v8bf*)(Bg);
      __builtin_prefetch(Ag + 32, 0, 0);
      __builtin_prefetch(Bg + bstep, 0, 0);
    }

    // --- fragments from LDS (ds_load_b128 pairs)
    v8bf a0lo = *(const v8bf*)(&As[wr * 32 + lrow][kb0]);
    v8bf a0hi = *(const v8bf*)(&As[wr * 32 + lrow][kb1]);
    v8bf a1lo = *(const v8bf*)(&As[wr * 32 + 16 + lrow][kb0]);
    v8bf a1hi = *(const v8bf*)(&As[wr * 32 + 16 + lrow][kb1]);
    v8bf b0lo = *(const v8bf*)(&Bs[wc * 32 + lrow][kb0]);
    v8bf b0hi = *(const v8bf*)(&Bs[wc * 32 + lrow][kb1]);
    v8bf b1lo = *(const v8bf*)(&Bs[wc * 32 + 16 + lrow][kb0]);
    v8bf b1hi = *(const v8bf*)(&Bs[wc * 32 + 16 + lrow][kb1]);
    v16bf a0, a1, b0, b1;
    #pragma unroll
    for (int e = 0; e < 8; ++e) {
      a0[e] = a0lo[e]; a0[e + 8] = a0hi[e];
      a1[e] = a1lo[e]; a1[e + 8] = a1hi[e];
      b0[e] = b0lo[e]; b0[e + 8] = b0hi[e];
      b1[e] = b1lo[e]; b1[e + 8] = b1hi[e];
    }
    c00 = __builtin_amdgcn_wmma_f32_16x16x32_bf16(false, a0, false, b0,
                                                  (short)0, c00, false, false);
    c01 = __builtin_amdgcn_wmma_f32_16x16x32_bf16(false, a0, false, b1,
                                                  (short)0, c01, false, false);
    c10 = __builtin_amdgcn_wmma_f32_16x16x32_bf16(false, a1, false, b0,
                                                  (short)0, c10, false, false);
    c11 = __builtin_amdgcn_wmma_f32_16x16x32_bf16(false, a1, false, b1,
                                                  (short)0, c11, false, false);
    __syncthreads();
  }

  // --- epilogue: C/D layout => VGPR v holds row (v + 8*hf), col = lrow
  #pragma unroll
  for (int ri = 0; ri < 2; ++ri) {
    #pragma unroll
    for (int ci = 0; ci < 2; ++ci) {
      v8f acc = ri ? (ci ? c11 : c10) : (ci ? c01 : c00);
      #pragma unroll
      for (int v = 0; v < 8; ++v) {
        const int grow = m0 + wr * 32 + ri * 16 + hf * 8 + v;
        const int gcol = n0 + wc * 32 + ci * 16 + lrow;
        float val = acc[v];
        if (bias)  val += bias[gcol];
        if (gelu)  val = 0.5f * val * (1.0f + erff(val * 0.7071067811865475f));
        const long long cidx = cbase + (long long)grow * ldc
                                     + (long long)gcol * cstride;
        if (resid) val += resid[cidx];
        if (Cf) Cf[cidx] = val;
        if (Cb) Cb[cidx] = f2bf(val);
      }
    }
  }
}

// ---------------------------------------------------------------------------
// LayerNorm over dim=512, one block (256 threads) per row, bf16 output
// ---------------------------------------------------------------------------
__global__ __launch_bounds__(256)
void layernorm_bf16(const float* __restrict__ x, const float* __restrict__ g,
                    const float* __restrict__ bta, bf16_t* __restrict__ out)
{
  __shared__ float red[256];
  const int row = blockIdx.x, tid = threadIdx.x;
  const long long base = (long long)row * 512;
  const float v0 = x[base + tid];
  const float v1 = x[base + tid + 256];
  red[tid] = v0 + v1;
  __syncthreads();
  for (int o = 128; o > 0; o >>= 1) { if (tid < o) red[tid] += red[tid + o]; __syncthreads(); }
  const float mean = red[0] * (1.0f / 512.0f);
  __syncthreads();
  red[tid] = (v0 - mean) * (v0 - mean) + (v1 - mean) * (v1 - mean);
  __syncthreads();
  for (int o = 128; o > 0; o >>= 1) { if (tid < o) red[tid] += red[tid + o]; __syncthreads(); }
  const float rstd = rsqrtf(red[0] * (1.0f / 512.0f) + 1e-5f);
  out[base + tid]       = f2bf((v0 - mean) * rstd * g[tid]       + bta[tid]);
  out[base + tid + 256] = f2bf((v1 - mean) * rstd * g[tid + 256] + bta[tid + 256]);
}

// ---------------------------------------------------------------------------
// Softmax over HEADS (dim=-1 of (q,k,b,h)) with fused Transformer-XL
// rel_shift on BD:  out[i,j] = j<=i ? BD[i, j+N-1-i] : (j==i+1 ? 0 :
//                                                       BD[i+1, j-i-2])
// Buffers are laid out (b,h,i,j), contiguous NxN per (b,h).
// ---------------------------------------------------------------------------
__global__ __launch_bounds__(256)
void softmax_heads(const float* __restrict__ AC, const float* __restrict__ BD,
                   bf16_t* __restrict__ attnb, float* __restrict__ attnf)
{
  const long long t = (long long)blockIdx.x * 256 + threadIdx.x; // < 8*1024*1024
  const int j = (int)(t & 1023);
  const int i = (int)((t >> 10) & 1023);
  const int b = (int)(t >> 20);
  const int N = 1024;
  int si, sj; bool zero;
  if (j <= i)           { si = i;     sj = j + N - 1 - i; zero = false; }
  else if (j == i + 1)  { si = 0;     sj = 0;             zero = true;  }
  else                  { si = i + 1; sj = j - i - 2;     zero = false; }

  float e[8]; float m = -1e30f;
  #pragma unroll
  for (int h = 0; h < 8; ++h) {
    const long long base = ((long long)(b * 8 + h)) << 20;
    const float ac = AC[base + ((long long)i << 10) + j];
    const float bd = zero ? 0.0f : BD[base + ((long long)si << 10) + sj];
    const float v = (ac + bd) * 0.125f;          // SCALE = 64^-0.5
    e[h] = v; m = fmaxf(m, v);
  }
  float s = 0.0f;
  #pragma unroll
  for (int h = 0; h < 8; ++h) { e[h] = __expf(e[h] - m); s += e[h]; }
  const float inv = 1.0f / s;
  #pragma unroll
  for (int h = 0; h < 8; ++h) {
    const long long idx = (((long long)(b * 8 + h)) << 20)
                        + ((long long)i << 10) + j;
    const float a = e[h] * inv;
    attnb[idx] = f2bf(a);
    if (attnf) attnf[idx] = a;   // last layer: straight into d_out (b,h,q,k)
  }
}

// ---------------------------------------------------------------------------
// Utility kernels
// ---------------------------------------------------------------------------
__global__ void cvt_f32_bf16(const float* __restrict__ in,
                             bf16_t* __restrict__ out, long long n)
{
  const long long i = (long long)blockIdx.x * 256 + threadIdx.x;
  if (i < n) out[i] = f2bf(in[i]);
}

__global__ void copy_f32(const float* __restrict__ in,
                         float* __restrict__ out, long long n)
{
  const long long i = (long long)blockIdx.x * 256 + threadIdx.x;
  if (i < n) out[i] = in[i];
}

// ---------------------------------------------------------------------------
// Host launcher
// ---------------------------------------------------------------------------
extern "C" void kernel_launch(void* const* d_in, const int* in_sizes, int n_in,
                              void* d_out, int out_size, void* d_ws, size_t ws_size,
                              hipStream_t stream)
{
  (void)in_sizes; (void)n_in; (void)out_size; (void)ws_size;

  const int DEPTH = 4, Bsz = 8, Nseq = 1024, DIM = 512, HEADS = 8, DHEAD = 64;
  const int HD = 512, FF = 2048;
  const long long MR  = (long long)Bsz * Nseq;              // 8192 rows
  const long long SNN = (long long)Nseq * Nseq;             // 1M
  const long long SCORES = (long long)Bsz * HEADS * SNN;    // 64M elems

  // inputs (setup_inputs order)
  const float* x_in    = (const float*)d_in[0];
  const float* r_t     = (const float*)d_in[1];
  const float* bias_pf = (const float*)d_in[3];
  const float* ln1_g   = (const float*)d_in[4];
  const float* ln1_b   = (const float*)d_in[5];
  const float* Wqkv    = (const float*)d_in[6];
  const float* Wkr_t   = (const float*)d_in[7];
  const float* Wout    = (const float*)d_in[9];
  const float* bout    = (const float*)d_in[10];
  const float* ln2_g   = (const float*)d_in[11];
  const float* ln2_b   = (const float*)d_in[12];
  const float* W1      = (const float*)d_in[13];
  const float* b1p     = (const float*)d_in[14];
  const float* W2      = (const float*)d_in[15];
  const float* b2p     = (const float*)d_in[16];
  // d_in[2] (r_p) and d_in[8] (Wkr_p) are computed-then-discarded in the ref.

  // workspace carving
  char* p = (char*)d_ws;
  auto carve = [&](long long bytes) -> char* {
    char* r = p; p += (bytes + 255) & ~255LL; return r;
  };
  bf16_t* rt_b   = (bf16_t*)carve(MR * DIM * 2);
  bf16_t* wqkv_b = (bf16_t*)carve((long long)DEPTH * DIM * 3 * HD * 2);
  bf16_t* wkr_b  = (bf16_t*)carve((long long)DEPTH * DIM * HD * 2);
  bf16_t* wout_b = (bf16_t*)carve((long long)DEPTH * HD * DIM * 2);
  bf16_t* w1_b   = (bf16_t*)carve((long long)DEPTH * DIM * FF * 2);
  bf16_t* w2_b   = (bf16_t*)carve((long long)DEPTH * FF * DIM * 2);
  float*  xbuf   = (float* )carve(MR * DIM * 4);
  bf16_t* xn_b   = (bf16_t*)carve(MR * DIM * 2);
  bf16_t* qb_b   = (bf16_t*)carve(MR * HD * 2);
  bf16_t* k_b    = (bf16_t*)carve(MR * HD * 2);
  bf16_t* v_b    = (bf16_t*)carve(MR * HD * 2);
  bf16_t* kr_b   = (bf16_t*)carve(MR * HD * 2);
  bf16_t* ao_b   = (bf16_t*)carve(MR * HD * 2);
  bf16_t* h_b    = (bf16_t*)carve(MR * FF * 2);
  float*  S_AC   = (float* )carve(SCORES * 4);
  float*  S_BD   = (float* )carve(SCORES * 4);
  bf16_t* attn_b = (bf16_t*)carve(SCORES * 2);

  auto cvt = [&](const float* src, bf16_t* dst, long long n) {
    cvt_f32_bf16<<<dim3((unsigned)((n + 255) / 256)), dim3(256), 0, stream>>>(src, dst, n);
  };
  // one-time conversions to bf16
  cvt(r_t,   rt_b,   MR * DIM);
  cvt(Wqkv,  wqkv_b, (long long)DEPTH * DIM * 3 * HD);
  cvt(Wkr_t, wkr_b,  (long long)DEPTH * DIM * HD);
  cvt(Wout,  wout_b, (long long)DEPTH * HD * DIM);
  cvt(W1,    w1_b,   (long long)DEPTH * DIM * FF);
  cvt(W2,    w2_b,   (long long)DEPTH * FF * DIM);
  copy_f32<<<dim3((unsigned)((MR * DIM + 255) / 256)), dim3(256), 0, stream>>>(
      x_in, xbuf, MR * DIM);

  auto gemm = [&](bool bt,
                  const bf16_t* A, int lda, long long Ab, long long Ah,
                  const bf16_t* Bp, int ldb, long long Bb, long long Bh,
                  float* Cf, bf16_t* Cb, int ldc, int cstride,
                  long long Cbs, long long Chs,
                  const float* bias, const float* resid, int gelu,
                  int M, int Nn, int K, int batches) {
    dim3 g((unsigned)(Nn / 64), (unsigned)(M / 128), (unsigned)batches), blk(256);
    if (bt)
      gemm_wmma_bf16<true><<<g, blk, 0, stream>>>(A, lda, Ab, Ah, Bp, ldb, Bb, Bh,
          Cf, Cb, ldc, cstride, Cbs, Chs, bias, resid, gelu, K);
    else
      gemm_wmma_bf16<false><<<g, blk, 0, stream>>>(A, lda, Ab, Ah, Bp, ldb, Bb, Bh,
          Cf, Cb, ldc, cstride, Cbs, Chs, bias, resid, gelu, K);
  };

  const long long rowHD = (long long)Nseq * HD;   // per-b stride of (b,n,hd)

  for (int d = 0; d < DEPTH; ++d) {
    const bf16_t* wqkvL = wqkv_b + (long long)d * DIM * 3 * HD;
    // ---- LN1 -> xn (bf16)
    layernorm_bf16<<<dim3((unsigned)MR), dim3(256), 0, stream>>>(
        xbuf, ln1_g + d * DIM, ln1_b + d * DIM, xn_b);
    // ---- q (+bias_pf), k, v projections (column slices of Wqkv)
    gemm(false, xn_b, DIM, 0, 0, wqkvL,          3 * HD, 0, 0,
         nullptr, qb_b, HD, 1, 0, 0, bias_pf, nullptr, 0, (int)MR, HD, DIM, 1);
    gemm(false, xn_b, DIM, 0, 0, wqkvL + HD,     3 * HD, 0, 0,
         nullptr, k_b,  HD, 1, 0, 0, nullptr, nullptr, 0, (int)MR, HD, DIM, 1);
    gemm(false, xn_b, DIM, 0, 0, wqkvL + 2 * HD, 3 * HD, 0, 0,
         nullptr, v_b,  HD, 1, 0, 0, nullptr, nullptr, 0, (int)MR, HD, DIM, 1);
    // ---- kr_t = r_t @ Wkr_t
    gemm(false, rt_b, DIM, 0, 0, wkr_b + (long long)d * DIM * HD, HD, 0, 0,
         nullptr, kr_b, HD, 1, 0, 0, nullptr, nullptr, 0, (int)MR, HD, DIM, 1);
    // ---- AC = qb . k^T  per (b,h)  -> (b,h,i,j) fp32
    gemm(true,  qb_b, HD, rowHD, DHEAD,  k_b,  HD, rowHD, DHEAD,
         S_AC, nullptr, Nseq, 1, (long long)HEADS * SNN, SNN,
         nullptr, nullptr, 0, Nseq, Nseq, DHEAD, Bsz * HEADS);
    // ---- BD = qb . kr_t^T per (b,h)
    gemm(true,  qb_b, HD, rowHD, DHEAD,  kr_b, HD, rowHD, DHEAD,
         S_BD, nullptr, Nseq, 1, (long long)HEADS * SNN, SNN,
         nullptr, nullptr, 0, Nseq, Nseq, DHEAD, Bsz * HEADS);
    // ---- softmax over heads with fused rel_shift
    float* attnf = (d == DEPTH - 1) ? ((float*)d_out + MR * DIM) : nullptr;
    softmax_heads<<<dim3((unsigned)((Bsz * SNN) / 256)), dim3(256), 0, stream>>>(
        S_AC, S_BD, attn_b, attnf);
    // ---- attn_out[b,i,h,:] = attn[b,h,i,:] @ v[b,:,h,:]
    gemm(false, attn_b, Nseq, (long long)HEADS * SNN, SNN,
         v_b, HD, rowHD, DHEAD,
         nullptr, ao_b, HD, 1, rowHD, DHEAD,
         nullptr, nullptr, 0, Nseq, DHEAD, Nseq, Bsz * HEADS);
    // ---- out projection + bout + residual -> x (fp32, in place)
    gemm(false, ao_b, HD, 0, 0, wout_b + (long long)d * HD * DIM, DIM, 0, 0,
         xbuf, nullptr, DIM, 1, 0, 0, bout + d * DIM, xbuf, 0,
         (int)MR, DIM, HD, 1);
    // ---- LN2 -> xn2 (reuse xn_b)
    layernorm_bf16<<<dim3((unsigned)MR), dim3(256), 0, stream>>>(
        xbuf, ln2_g + d * DIM, ln2_b + d * DIM, xn_b);
    // ---- FFN: h = gelu(xn2 @ W1 + b1) ; x = h @ W2 + b2 + x
    gemm(false, xn_b, DIM, 0, 0, w1_b + (long long)d * DIM * FF, FF, 0, 0,
         nullptr, h_b, FF, 1, 0, 0, b1p + d * FF, nullptr, 1,
         (int)MR, FF, DIM, 1);
    gemm(false, h_b, FF, 0, 0, w2_b + (long long)d * FF * DIM, DIM, 0, 0,
         xbuf, nullptr, DIM, 1, 0, 0, b2p + d * DIM, xbuf, 0,
         (int)MR, DIM, FF, 1);
  }

  // final x -> d_out[0 : MR*DIM]
  copy_f32<<<dim3((unsigned)((MR * DIM + 255) / 256)), dim3(256), 0, stream>>>(
      xbuf, (float*)d_out, MR * DIM);
}
